// GeometricGuidanceNetwork_81990925681352
// MI455X (gfx1250) — compile-verified, matrix operand order checked by reading, not run on hardware
//
#include <hip/hip_runtime.h>
#include <hip/hip_bf16.h>
#include <math.h>

typedef __attribute__((ext_vector_type(16))) __bf16 v16bf;
typedef __attribute__((ext_vector_type(8)))  __bf16 v8bf;
typedef __attribute__((ext_vector_type(8)))  float  v8f;
typedef __attribute__((ext_vector_type(4)))  unsigned int u32x4;
typedef __attribute__((ext_vector_type(4)))  int          i32x4;
typedef __attribute__((ext_vector_type(8)))  int          i32x8;

#define N_NODES 50000
#define N_EDGES 800000
#define NB      64
#define ATOMF   100
#define HID     128
#define NLAYERS 4
#define TD      64

__device__ __forceinline__ float siluf(float x) {
  return x * __builtin_amdgcn_rcpf(1.0f + __expf(-x));
}

// ---------------------------------------------------------------------------
// TDM: 1-D bf16 bulk copy global -> LDS (D# per CDNA5 ISA)
//   group0: [1:0]=count=1, [63:32]=lds_addr, [120:64]=global_addr, [127:126]=2
//   group1: [17:16]=data_size(1 => 2B), tensor_dim0=tile_dim0=nelem (1-D tile)
//   groups 2/3 unused (2-D descriptor) -> zero-filled; cpol=0
// ---------------------------------------------------------------------------
__device__ __forceinline__ void tdm_load_bf16_1d(unsigned lds_off, const void* gptr,
                                                 unsigned nelem) {
  unsigned long long ga = (unsigned long long)gptr;
  u32x4 g0;
  g0[0] = 1u;                                            // count=1, no gather
  g0[1] = lds_off;                                       // LDS byte address
  g0[2] = (unsigned)ga;                                  // global addr [31:0]
  g0[3] = (unsigned)((ga >> 32) & 0x1FFFFFFu) | (2u << 30); // addr[56:32] | type=2
  i32x8 g1;
  g1[0] = 1 << 16;                                       // data_size = 2 bytes
  g1[1] = (int)(nelem << 16);                            // tensor_dim0[15:0] @ [63:48]
  g1[2] = (int)((nelem >> 16) & 0xFFFFu) | (1 << 16);    // tensor_dim0[31:16], tensor_dim1=1
  g1[3] = (int)(nelem << 16);                            // tile_dim0 @ [127:112]
  g1[4] = 0;                                             // tile_dim1=0 (unused)
  g1[5] = (int)nelem;                                    // tensor_dim0_stride lo32
  g1[6] = 0;
  g1[7] = 0;
  i32x4 z4 = {0, 0, 0, 0};
  i32x8 z8 = {0, 0, 0, 0, 0, 0, 0, 0};
  __builtin_amdgcn_tensor_load_to_lds(g0, g1, z4, z4, z8, 0);
}

// ---------------------------------------------------------------------------
// time embedding: sinusoidal -> Lin-SiLU-Lin -> proj(64->128)   (B=64, tiny)
// ---------------------------------------------------------------------------
__global__ void k_time_embed(const float* __restrict__ t,
                             const float* __restrict__ te1w, const float* __restrict__ te1b,
                             const float* __restrict__ te2w, const float* __restrict__ te2b,
                             const float* __restrict__ tpw,  const float* __restrict__ tpb,
                             float* __restrict__ t_emb) {
  __shared__ float e0[NB][TD];
  __shared__ float e1[NB][TD];
  int b = threadIdx.x;
  if (b >= NB) return;
  float tv = t[b];
  const float c = 9.210340371976184f / 32.0f; // ln(10000)/half
  for (int k = 0; k < 32; ++k) {
    float inv = __expf(-c * (float)k);
    float s = tv * inv;
    e0[b][k]      = __sinf(s);
    e0[b][32 + k] = __cosf(s);
  }
  for (int j = 0; j < TD; ++j) {
    float a = te1b[j];
    for (int k = 0; k < TD; ++k) a += e0[b][k] * te1w[k * TD + j];
    e1[b][j] = siluf(a);
  }
  for (int j = 0; j < TD; ++j) {
    float a = te2b[j];
    for (int k = 0; k < TD; ++k) a += e1[b][k] * te2w[k * TD + j];
    e0[b][j] = a; // reuse
  }
  for (int f = 0; f < HID; ++f) {
    float a = tpb[f];
    for (int k = 0; k < TD; ++k) a += e0[b][k] * tpw[k * HID + f];
    t_emb[b * HID + f] = a;
  }
}

// ---------------------------------------------------------------------------
// node init: softmax(theta) @ W_node_in + b + t_emb[batch]   (1 block / node)
// ---------------------------------------------------------------------------
__global__ void __launch_bounds__(128) k_node_init(const float* __restrict__ theta,
                            const int* __restrict__ batch,
                            const float* __restrict__ W, const float* __restrict__ bias,
                            const float* __restrict__ t_emb, float* __restrict__ x) {
  __shared__ float p[ATOMF];
  __shared__ float red[128];
  int n = blockIdx.x, tid = threadIdx.x;
  float v = (tid < ATOMF) ? theta[(size_t)n * ATOMF + tid] : -1e30f;
  red[tid] = v; __syncthreads();
  for (int s = 64; s > 0; s >>= 1) { if (tid < s) red[tid] = fmaxf(red[tid], red[tid + s]); __syncthreads(); }
  float mx = red[0]; __syncthreads();
  float ex = (tid < ATOMF) ? __expf(v - mx) : 0.0f;
  red[tid] = ex; __syncthreads();
  for (int s = 64; s > 0; s >>= 1) { if (tid < s) red[tid] += red[tid + s]; __syncthreads(); }
  float denom = red[0];
  if (tid < ATOMF) p[tid] = ex / denom;
  __syncthreads();
  float acc = bias[tid];
  for (int k = 0; k < ATOMF; ++k) acc += p[k] * W[k * HID + tid];
  int g = batch[n];
  x[(size_t)n * HID + tid] = acc + t_emb[g * HID + tid];
}

// ---------------------------------------------------------------------------
// edge features: dist MLP + dir MLP + LayerNorm  (2 edges / 128-thread block)
// ---------------------------------------------------------------------------
__global__ void __launch_bounds__(128) k_edge_feat(const float* __restrict__ pos,
                            const int* __restrict__ row, const int* __restrict__ col,
                            const float* __restrict__ d1w, const float* __restrict__ d1b,
                            const float* __restrict__ d2w, const float* __restrict__ d2b,
                            const float* __restrict__ r1w, const float* __restrict__ r1b,
                            const float* __restrict__ r2w, const float* __restrict__ r2b,
                            const float* __restrict__ lng, const float* __restrict__ lnb,
                            float* __restrict__ ef) {
  __shared__ float h1[2][64];
  __shared__ float h2[2][64];
  __shared__ float out[2][128];
  __shared__ float red[2][64];
  int sub = threadIdx.x >> 6;
  int j = threadIdx.x & 63;
  size_t e = (size_t)blockIdx.x * 2 + sub;
  int r = row[e], c = col[e];
  float vx = pos[c * 3 + 0] - pos[r * 3 + 0];
  float vy = pos[c * 3 + 1] - pos[r * 3 + 1];
  float vz = pos[c * 3 + 2] - pos[r * 3 + 2];
  float dist = sqrtf(vx * vx + vy * vy + vz * vz);
  float inv = 1.0f / (dist + 1e-8f);
  float dx = vx * inv, dy = vy * inv, dz = vz * inv;
  h1[sub][j] = siluf(dist * d1w[j] + d1b[j]);
  h2[sub][j] = siluf(dx * r1w[j] + dy * r1w[64 + j] + dz * r1w[128 + j] + r1b[j]);
  __syncthreads();
  float a1 = d2b[j], a2 = r2b[j];
  for (int k = 0; k < 64; ++k) {
    a1 += h1[sub][k] * d2w[k * 64 + j];
    a2 += h2[sub][k] * r2w[k * 64 + j];
  }
  out[sub][j] = a1; out[sub][64 + j] = a2;
  red[sub][j] = a1 + a2; __syncthreads();
  for (int s = 32; s > 0; s >>= 1) { if (j < s) red[sub][j] += red[sub][j + s]; __syncthreads(); }
  float mu = red[sub][0] * (1.0f / 128.0f); __syncthreads();
  float dv1 = out[sub][j] - mu, dv2 = out[sub][64 + j] - mu;
  red[sub][j] = dv1 * dv1 + dv2 * dv2; __syncthreads();
  for (int s = 32; s > 0; s >>= 1) { if (j < s) red[sub][j] += red[sub][j + s]; __syncthreads(); }
  float rstd = rsqrtf(red[sub][0] * (1.0f / 128.0f) + 1e-5f);
  ef[e * HID + j]      = dv1 * rstd * lng[j]      + lnb[j];
  ef[e * HID + 64 + j] = dv2 * rstd * lng[64 + j] + lnb[64 + j];
}

// ---------------------------------------------------------------------------
// weight prep: f32 [K][N] -> bf16 transposed [N][K]
// ---------------------------------------------------------------------------
__global__ void k_prep_wt(const float* __restrict__ W, unsigned short* __restrict__ Wt,
                          int K, int Ncols) {
  int idx = blockIdx.x * 256 + threadIdx.x;
  if (idx >= K * Ncols) return;
  int k = idx / Ncols, n = idx % Ncols;
  __bf16 v = (__bf16)W[idx];
  Wt[n * K + k] = *(unsigned short*)&v;
}

// ---------------------------------------------------------------------------
// message MLP (WMMA bf16): per 16-edge tile
//   m_ij = SiLU([x[row]|x[col]|ef] @ W1 + b1) @ W2 + b2 ; atomic scatter to row
// Weights staged in LDS via Tensor Data Mover; B-fragments read with ds_load.
// Dynamic LDS: [0,96K) = W1t bf16 [128][384], [96K,128K) = W2t bf16 [128][128],
//              [128K,144K) = per-wave h tiles [4][16][128] bf16
// ---------------------------------------------------------------------------
__global__ void __launch_bounds__(128) k_message(const float* __restrict__ x,
                          const float* __restrict__ ef,
                          const int* __restrict__ row, const int* __restrict__ col,
                          const unsigned short* __restrict__ W1t_,  // [128][384] bf16
                          const unsigned short* __restrict__ W2t_,  // [128][128] bf16
                          const float* __restrict__ b1, const float* __restrict__ b2,
                          float* __restrict__ mi) {
  extern __shared__ char smem[];
  __bf16* W1l = (__bf16*)smem;                    // 98304 B
  __bf16* W2l = (__bf16*)(smem + 98304);          // 32768 B
  int lane = threadIdx.x & 31;
  int wave = threadIdx.x >> 5;
  __bf16 (*hb)[HID] = ((__bf16(*)[HID])(smem + 131072)) + wave * 16;
  int m = lane & 15;   // A-row (edge in tile) == B/C/D column-within-Ntile
  int g = lane >> 4;   // half-wave

  // stage both weight matrices into LDS with the TDM, once per block
  if (wave == 0) {
    tdm_load_bf16_1d((unsigned)(unsigned long long)(void*)W1l, W1t_, 384u * 128u);
    tdm_load_bf16_1d((unsigned)(unsigned long long)(void*)W2l, W2t_, 128u * 128u);
    __builtin_amdgcn_s_wait_tensorcnt(0);
  }
  __syncthreads();

  // tile-invariant bias broadcasts
  float bias1[8], bias2[8];
#pragma unroll
  for (int n = 0; n < 8; ++n) { bias1[n] = b1[n * 16 + m]; bias2[n] = b2[n * 16 + m]; }

  int wgid = blockIdx.x * 4 + wave;
  int nwtot = gridDim.x * 4;
  const int ntiles = N_EDGES / 16;

  for (int tile = wgid; tile < ntiles; tile += nwtot) {
    int et = tile * 16;
    int r = row[et + m];
    int c = col[et + m];
    const float* xr  = x  + (size_t)r * HID;
    const float* xc  = x  + (size_t)c * HID;
    const float* efp = ef + (size_t)(et + m) * HID;
    if (tile + nwtot < ntiles)
      __builtin_prefetch(ef + (size_t)((tile + nwtot) * 16 + m) * HID, 0, 1);

    // ---- GEMM1: [16 x 384] @ [384 x 128] -----------------------------------
    v8f acc[8];
#pragma unroll
    for (int n = 0; n < 8; ++n) {
#pragma unroll
      for (int i = 0; i < 8; ++i) acc[n][i] = bias1[n];
    }
    for (int cch = 0; cch < 12; ++cch) {
      const float* src = (cch < 4) ? xr : (cch < 8) ? xc : efp;
      int fk = (cch & 3) * 32 + g * 8;
      float4 f0 = *(const float4*)(src + fk);
      float4 f1 = *(const float4*)(src + fk + 4);
      float4 f2 = *(const float4*)(src + fk + 16);
      float4 f3 = *(const float4*)(src + fk + 20);
      v16bf a;
      a[0]=(__bf16)f0.x;  a[1]=(__bf16)f0.y;  a[2]=(__bf16)f0.z;  a[3]=(__bf16)f0.w;
      a[4]=(__bf16)f1.x;  a[5]=(__bf16)f1.y;  a[6]=(__bf16)f1.z;  a[7]=(__bf16)f1.w;
      a[8]=(__bf16)f2.x;  a[9]=(__bf16)f2.y;  a[10]=(__bf16)f2.z; a[11]=(__bf16)f2.w;
      a[12]=(__bf16)f3.x; a[13]=(__bf16)f3.y; a[14]=(__bf16)f3.z; a[15]=(__bf16)f3.w;
#pragma unroll
      for (int n = 0; n < 8; ++n) {
        const __bf16* wp = W1l + (unsigned)(n * 16 + m) * 384 + cch * 32 + g * 8;
        v8bf blo = *(const v8bf*)wp;
        v8bf bhi = *(const v8bf*)(wp + 16);
        v16bf bf;
#pragma unroll
        for (int i = 0; i < 8; ++i) { bf[i] = blo[i]; bf[i + 8] = bhi[i]; }
        acc[n] = __builtin_amdgcn_wmma_f32_16x16x32_bf16(
            false, a, false, bf, (short)0, acc[n], false, false);
      }
    }
    // SiLU + stage h tile in LDS (D layout: elem v -> row v+8*g, col n*16+m)
#pragma unroll
    for (int n = 0; n < 8; ++n)
#pragma unroll
      for (int v = 0; v < 8; ++v)
        hb[v + 8 * g][n * 16 + m] = (__bf16)siluf(acc[n][v]);
    asm volatile("s_wait_dscnt 0x0" ::: "memory"); // cross-lane LDS RAW

    // ---- GEMM2: [16 x 128] @ [128 x 128] -----------------------------------
    v8f acc2[8];
#pragma unroll
    for (int n = 0; n < 8; ++n) {
#pragma unroll
      for (int i = 0; i < 8; ++i) acc2[n][i] = bias2[n];
    }
#pragma unroll
    for (int c2 = 0; c2 < 4; ++c2) {
      const __bf16* hp = &hb[m][c2 * 32 + g * 8];
      v8bf alo = *(const v8bf*)hp;
      v8bf ahi = *(const v8bf*)(hp + 16);
      v16bf a;
#pragma unroll
      for (int i = 0; i < 8; ++i) { a[i] = alo[i]; a[i + 8] = ahi[i]; }
#pragma unroll
      for (int n = 0; n < 8; ++n) {
        const __bf16* wp = W2l + (unsigned)(n * 16 + m) * 128 + c2 * 32 + g * 8;
        v8bf blo = *(const v8bf*)wp;
        v8bf bhi = *(const v8bf*)(wp + 16);
        v16bf bf;
#pragma unroll
        for (int i = 0; i < 8; ++i) { bf[i] = blo[i]; bf[i + 8] = bhi[i]; }
        acc2[n] = __builtin_amdgcn_wmma_f32_16x16x32_bf16(
            false, a, false, bf, (short)0, acc2[n], false, false);
      }
    }
    // scatter-add m_ij into m_i[row]
#pragma unroll
    for (int v = 0; v < 8; ++v) {
      int rv = row[et + v + 8 * g];
      float* dst = mi + (size_t)rv * HID;
#pragma unroll
      for (int n = 0; n < 8; ++n)
        unsafeAtomicAdd(dst + n * 16 + m, acc2[n][v]);
    }
  }
}

// ---------------------------------------------------------------------------
// node update: x = LN(x+mi); x = LN(x + FFN(x))   (1 block / node)
// ---------------------------------------------------------------------------
__global__ void __launch_bounds__(128) k_node_update(float* __restrict__ x,
                          const float* __restrict__ mi,
                          const float* __restrict__ g1, const float* __restrict__ bb1,
                          const float* __restrict__ f1w, const float* __restrict__ f1b,
                          const float* __restrict__ f2w, const float* __restrict__ f2b,
                          const float* __restrict__ g2, const float* __restrict__ bb2) {
  __shared__ float xs[HID];
  __shared__ float hs[2 * HID];
  __shared__ float red[128];
  int n = blockIdx.x, f = threadIdx.x;
  float v = x[(size_t)n * HID + f] + mi[(size_t)n * HID + f];
  red[f] = v; __syncthreads();
  for (int s = 64; s > 0; s >>= 1) { if (f < s) red[f] += red[f + s]; __syncthreads(); }
  float mu = red[0] * (1.0f / 128.0f); __syncthreads();
  float d = v - mu; red[f] = d * d; __syncthreads();
  for (int s = 64; s > 0; s >>= 1) { if (f < s) red[f] += red[f + s]; __syncthreads(); }
  float x1 = d * rsqrtf(red[0] * (1.0f / 128.0f) + 1e-5f) * g1[f] + bb1[f];
  __syncthreads();
  xs[f] = x1; __syncthreads();
  for (int jj = 0; jj < 2; ++jj) {
    int j = f + jj * 128;
    float a = f1b[j];
    for (int k = 0; k < HID; ++k) a += xs[k] * f1w[k * 256 + j];
    hs[j] = siluf(a);
  }
  __syncthreads();
  float a = f2b[f];
  for (int j = 0; j < 256; ++j) a += hs[j] * f2w[j * 128 + f];
  float v2 = x1 + a;
  __syncthreads();
  red[f] = v2; __syncthreads();
  for (int s = 64; s > 0; s >>= 1) { if (f < s) red[f] += red[f + s]; __syncthreads(); }
  float mu2 = red[0] * (1.0f / 128.0f); __syncthreads();
  float d2 = v2 - mu2; red[f] = d2 * d2; __syncthreads();
  for (int s = 64; s > 0; s >>= 1) { if (f < s) red[f] += red[f + s]; __syncthreads(); }
  x[(size_t)n * HID + f] = d2 * rsqrtf(red[0] * (1.0f / 128.0f) + 1e-5f) * g2[f] + bb2[f];
}

// ---------------------------------------------------------------------------
// pooling: batch is sorted -> binary-search range per graph
// ---------------------------------------------------------------------------
__global__ void __launch_bounds__(128) k_pool(const float* __restrict__ x,
                       const int* __restrict__ batch,
                       float* __restrict__ meanp, float* __restrict__ maxp) {
  __shared__ int se[2];
  int gi = blockIdx.x, f = threadIdx.x;
  if (f < 2) {
    int tgt = gi + f;
    int lo = 0, hi = N_NODES;
    while (lo < hi) { int mid = (lo + hi) >> 1; if (batch[mid] < tgt) lo = mid + 1; else hi = mid; }
    se[f] = lo;
  }
  __syncthreads();
  int s = se[0], e = se[1];
  float sum = 0.0f, mx = -3.0e38f;
  for (int nn = s; nn < e; ++nn) {
    float v = x[(size_t)nn * HID + f];
    sum += v; mx = fmaxf(mx, v);
  }
  float cnt = (float)(e - s);
  meanp[gi * HID + f] = sum / fmaxf(cnt, 1.0f);
  maxp[gi * HID + f]  = mx;
}

// ---------------------------------------------------------------------------
// heads: pool proj + mu / sigma MLPs  (1 block / graph)
// ---------------------------------------------------------------------------
__global__ void __launch_bounds__(128) k_head(const float* __restrict__ meanp,
                       const float* __restrict__ maxp,
                       const float* __restrict__ pw, const float* __restrict__ pb,
                       const float* __restrict__ m1w, const float* __restrict__ m1b,
                       const float* __restrict__ m2w, const float* __restrict__ m2b,
                       const float* __restrict__ s1w, const float* __restrict__ s1b,
                       const float* __restrict__ s2w, const float* __restrict__ s2b,
                       float* __restrict__ out) {
  __shared__ float cat[256];
  __shared__ float gf[HID];
  __shared__ float h[HID];
  int gi = blockIdx.x, f = threadIdx.x;
  cat[f] = meanp[gi * HID + f];
  cat[128 + f] = maxp[gi * HID + f];
  __syncthreads();
  float a = pb[f];
  for (int k = 0; k < 256; ++k) a += cat[k] * pw[k * HID + f];
  gf[f] = siluf(a); __syncthreads();
  a = m1b[f];
  for (int k = 0; k < HID; ++k) a += gf[k] * m1w[k * HID + f];
  h[f] = siluf(a); __syncthreads();
  if (f < 2) {
    float mv = m2b[f];
    for (int k = 0; k < HID; ++k) mv += h[k] * m2w[k * 2 + f];
    out[gi * 2 + f] = 1.0f / (1.0f + __expf(-mv));
  }
  __syncthreads();
  a = s1b[f];
  for (int k = 0; k < HID; ++k) a += gf[k] * s1w[k * HID + f];
  h[f] = siluf(a); __syncthreads();
  if (f < 2) {
    float rv = s2b[f];
    for (int k = 0; k < HID; ++k) rv += h[k] * s2w[k * 2 + f];
    float sp = (rv > 20.0f) ? rv : log1pf(__expf(rv));
    float sg = sp + 1e-3f;
    out[NB * 2 + gi * 2 + f] = fminf(fmaxf(sg, 1e-3f), 0.08f);
  }
}

// ---------------------------------------------------------------------------
// host launch
// ---------------------------------------------------------------------------
// params flattened (jax pytree order, dict keys sorted):
//   5.. : blocks[L]{ffn1 W,b ffn2 W,b ln1 g,b ln2 g,b msg1 W,b msg2 W,b} L=0..3
//   53: dir1W 54: dir1b 55: dir2W 56: dir2b 57: dist1W 58: dist1b 59: dist2W 60: dist2b
//   61: elnG 62: elnB 63: mu1W 64: mu1b 65: mu2W 66: mu2b 67: nodeW 68: nodeB
//   69: poolW 70: poolB 71: sig1W 72: sig1b 73: sig2W 74: sig2b
//   75: te1W 76: te1b 77: te2W 78: te2b 79: tprojW 80: tprojb
extern "C" void kernel_launch(void* const* d_in, const int* in_sizes, int n_in,
                              void* d_out, int out_size, void* d_ws, size_t ws_size,
                              hipStream_t stream) {
  const float* theta = (const float*)d_in[0];
  const float* pos   = (const float*)d_in[1];
  const float* t     = (const float*)d_in[2];
  const int*   batch = (const int*)d_in[3];
  const int*   eidx  = (const int*)d_in[4];
  const int*   row   = eidx;
  const int*   col   = eidx + N_EDGES;
  auto P = [&](int i) { return (const float*)d_in[i]; };

  // workspace carve (256B aligned)
  char* ws = (char*)d_ws;
  size_t off = 0;
  auto carve = [&](size_t bytes) { void* p = ws + off; off = (off + bytes + 255) & ~(size_t)255; return p; };
  float* x     = (float*)carve((size_t)N_NODES * HID * 4);
  float* mi    = (float*)carve((size_t)N_NODES * HID * 4);
  float* t_emb = (float*)carve((size_t)NB * HID * 4);
  float* ef    = (float*)carve((size_t)N_EDGES * HID * 4);
  unsigned short* W1t[NLAYERS];
  unsigned short* W2t[NLAYERS];
  for (int L = 0; L < NLAYERS; ++L) {
    W1t[L] = (unsigned short*)carve((size_t)384 * 128 * 2);
    W2t[L] = (unsigned short*)carve((size_t)128 * 128 * 2);
  }
  float* meanp = (float*)carve((size_t)NB * HID * 4);
  float* maxp  = (float*)carve((size_t)NB * HID * 4);

  k_time_embed<<<1, 64, 0, stream>>>(t, P(75), P(76), P(77), P(78), P(79), P(80), t_emb);
  k_node_init<<<N_NODES, 128, 0, stream>>>(theta, batch, P(67), P(68), t_emb, x);
  k_edge_feat<<<N_EDGES / 2, 128, 0, stream>>>(pos, row, col,
      P(57), P(58), P(59), P(60), P(53), P(54), P(55), P(56), P(61), P(62), ef);

  for (int L = 0; L < NLAYERS; ++L) {
    int base = 5 + L * 12;
    k_prep_wt<<<(384 * 128 + 255) / 256, 256, 0, stream>>>(P(base + 8),  W1t[L], 384, 128);
    k_prep_wt<<<(128 * 128 + 255) / 256, 256, 0, stream>>>(P(base + 10), W2t[L], 128, 128);
  }
  const size_t msg_lds = 98304 + 32768 + 4 * 16 * HID * 2; // W1 + W2 + h tiles = 147456
  for (int L = 0; L < NLAYERS; ++L) {
    int base = 5 + L * 12;
    (void)hipMemsetAsync(mi, 0, (size_t)N_NODES * HID * 4, stream);
    k_message<<<2048, 128, msg_lds, stream>>>(x, ef, row, col, W1t[L], W2t[L],
                                              P(base + 9), P(base + 11), mi);
    k_node_update<<<N_NODES, 128, 0, stream>>>(x, mi,
        P(base + 4), P(base + 5),             // ln1 g,b
        P(base + 0), P(base + 1),             // ffn1 W,b
        P(base + 2), P(base + 3),             // ffn2 W,b
        P(base + 6), P(base + 7));            // ln2 g,b
  }
  k_pool<<<NB, 128, 0, stream>>>(x, batch, meanp, maxp);
  k_head<<<NB, 128, 0, stream>>>(meanp, maxp, P(69), P(70),
                                 P(63), P(64), P(65), P(66),
                                 P(71), P(72), P(73), P(74), (float*)d_out);
}
